// WaggleGate_86835648790608
// MI455X (gfx1250) — matrix operand order repaired and checked by reading, not compiled, not run on hardware
//
#include <hip/hip_runtime.h>
#include <hip/hip_bf16.h>

#define B_TOK 16384
#define DDIM  1024
#define NEXP  6
#define HDIM  256
#define EPS_F 0.1f

typedef __attribute__((ext_vector_type(16))) __bf16 v16bf;
typedef __attribute__((ext_vector_type(8)))  __bf16 v8bf;
typedef __attribute__((ext_vector_type(8)))  float  v8f;

union AFrag { v16bf v; v8bf h[2]; };

// ---------------------------------------------------------------------------
// Zero the load-balance accumulator (ws is not re-poisoned between replays).
// ---------------------------------------------------------------------------
__global__ void init_kernel(float* __restrict__ loadsum) {
    if (threadIdx.x < NEXP) loadsum[threadIdx.x] = 0.0f;
}

// ---------------------------------------------------------------------------
// Convert x fp32 -> bf16 (row-major unchanged). 32 MB, stays L2-resident so
// the MoE kernel's A-fragment loads are L2 hits across all 6 expert passes.
// ---------------------------------------------------------------------------
__global__ __launch_bounds__(256)
void convert_x_kernel(const float* __restrict__ x, __bf16* __restrict__ xh) {
    size_t i = ((size_t)blockIdx.x * blockDim.x + threadIdx.x) * 4;
    float4 v = *(const float4*)(x + i);
    xh[i + 0] = (__bf16)v.x; xh[i + 1] = (__bf16)v.y;
    xh[i + 2] = (__bf16)v.z; xh[i + 3] = (__bf16)v.w;
}

// ---------------------------------------------------------------------------
// Convert expert weights fp32 -> bf16, transposed so WMMA B-fragments are
// contiguous:  W1t[e][h][d] = W1[e][d][h],  W2t[e][d][h] = W2[e][h][d].
// ---------------------------------------------------------------------------
__global__ void convert_wt_kernel(const float* __restrict__ W1,
                                  const float* __restrict__ W2,
                                  __bf16* __restrict__ W1t,
                                  __bf16* __restrict__ W2t) {
    int idx = blockIdx.x * blockDim.x + threadIdx.x;
    const int total = NEXP * DDIM * HDIM;
    if (idx >= total) return;
    {   // W1: [E][D][H] -> W1t: [E][H][D]
        int e = idx / (DDIM * HDIM);
        int r = idx % (DDIM * HDIM);
        int d = r / HDIM;
        int h = r % HDIM;
        W1t[(size_t)e * HDIM * DDIM + (size_t)h * DDIM + d] = (__bf16)W1[idx];
    }
    {   // W2: [E][H][D] -> W2t: [E][D][H]
        int e = idx / (HDIM * DDIM);
        int r = idx % (HDIM * DDIM);
        int h = r / DDIM;
        int d = r % DDIM;
        W2t[(size_t)e * DDIM * HDIM + (size_t)d * HDIM + h] = (__bf16)W2[idx];
    }
}

// ---------------------------------------------------------------------------
// Router: one wave32 per token. logits = x@Wr + br, softmax, eps-blend,
// top-2 -> dense gate matrix wcomb[B][E] (0 if expert not selected),
// accumulate blended probs into loadsum[E] for the aux loss.
// ---------------------------------------------------------------------------
__global__ __launch_bounds__(256)
void router_kernel(const float* __restrict__ x,
                   const float* __restrict__ Wr,
                   const float* __restrict__ br,
                   float* __restrict__ wcomb,
                   float* __restrict__ loadsum) {
    __shared__ float lsum[NEXP];
    const int tid  = threadIdx.x;
    const int lane = tid & 31;
    const int wave = tid >> 5;                 // 8 waves -> 8 tokens/block
    const int tok  = blockIdx.x * 8 + wave;

    if (tid < NEXP) lsum[tid] = 0.0f;
    __syncthreads();

    float part[NEXP];
    #pragma unroll
    for (int e = 0; e < NEXP; e++) part[e] = 0.0f;

    const float* xr = x + (size_t)tok * DDIM;
    for (int d = lane; d < DDIM; d += 32) {    // coalesced per-lane
        float xv = xr[d];
        const float* wr = Wr + (size_t)d * NEXP;
        #pragma unroll
        for (int e = 0; e < NEXP; e++) part[e] += xv * wr[e];
    }
    #pragma unroll
    for (int e = 0; e < NEXP; e++)
        #pragma unroll
        for (int off = 16; off > 0; off >>= 1)
            part[e] += __shfl_xor(part[e], off, 32);

    if (lane == 0) {
        float p[NEXP];
        float mx = -1e30f;
        #pragma unroll
        for (int e = 0; e < NEXP; e++) { p[e] = part[e] + br[e]; mx = fmaxf(mx, p[e]); }
        float s = 0.0f;
        #pragma unroll
        for (int e = 0; e < NEXP; e++) { p[e] = __expf(p[e] - mx); s += p[e]; }
        float inv = 1.0f / s;
        #pragma unroll
        for (int e = 0; e < NEXP; e++)
            p[e] = (1.0f - EPS_F) * p[e] * inv + EPS_F / (float)NEXP;

        // top-2 (first index wins ties, matching lax.top_k)
        int i1 = 0;
        #pragma unroll
        for (int e = 1; e < NEXP; e++) if (p[e] > p[i1]) i1 = e;
        int i2 = (i1 == 0) ? 1 : 0;
        #pragma unroll
        for (int e = 0; e < NEXP; e++) if (e != i1 && p[e] > p[i2]) i2 = e;

        float* wc = wcomb + (size_t)tok * NEXP;
        #pragma unroll
        for (int e = 0; e < NEXP; e++) {
            wc[e] = (e == i1 || e == i2) ? p[e] : 0.0f;
            atomicAdd(&lsum[e], p[e]);
        }
    }
    __syncthreads();
    if (tid < NEXP) atomicAdd(&loadsum[tid], lsum[tid]);
}

// ---------------------------------------------------------------------------
// Aux loss finalize -> out[B*D]
// ---------------------------------------------------------------------------
__global__ void aux_kernel(const float* __restrict__ loadsum,
                           float* __restrict__ out_aux) {
    if (threadIdx.x == 0) {
        float s = 0.0f;
        for (int e = 0; e < NEXP; e++) {
            float ld = loadsum[e] / (float)B_TOK;
            s += ld * logf(ld * (float)NEXP + 1e-9f);
        }
        out_aux[0] = s / logf((float)NEXP + 1e-9f);
    }
}

// ---------------------------------------------------------------------------
// Fused dense MoE expert kernel. Block = 64 tokens, 512 threads = 16 wave32s
// arranged as 4 M-groups x 4 N-groups (4x weight reuse vs 16-token tiles ->
// ~1.5 TB instead of 6 TB of L2 weight traffic; back to matrix-core bound).
//
// Per expert:
//   GEMM1: [64,1024]@[1024,256] (A from L2-resident bf16 xh) -> gelu * gate
//          -> hs (bf16, LDS). Gate folded into h so GEMM2's WMMA C-operand
//          natively accumulates the weighted expert mixture across experts.
//   GEMM2: [64,256]@[256,1024] accumulated into persistent v8f oacc[16].
// Epilogue adds the gated bias sum_e w_e * b2[e].
// ---------------------------------------------------------------------------
__global__ __launch_bounds__(512)
void moe_expert_kernel(const __bf16* __restrict__ xh,    // [B][D] bf16
                       const __bf16* __restrict__ W1t,   // [E][H][D]
                       const float*  __restrict__ b1,    // [E][H]
                       const __bf16* __restrict__ W2t,   // [E][D][H]
                       const float*  __restrict__ b2,    // [E][D]
                       const float*  __restrict__ wcomb, // [B][E]
                       float* __restrict__ out) {        // [B][D]
    __shared__ __bf16 hs[64 * HDIM];    // 32 KB hidden tile (gated, bf16)
    __shared__ float  wls[64 * NEXP];   // 1.5 KB gate weights

    const int tid  = threadIdx.x;
    const int lane = tid & 31;
    const int wave = tid >> 5;          // 0..15
    const int mg   = wave & 3;          // M-group: rows [mg*16, mg*16+16)
    const int ng   = wave >> 2;         // N-group
    const int tok0 = blockIdx.x * 64;

    if (tid < 64 * NEXP) wls[tid] = wcomb[(size_t)tok0 * NEXP + tid];
    __syncthreads();

    // Per-lane WMMA fragment geometry (wave32, 16x16x32 bf16):
    //   A: lane holds row (lane&15); K = hi16*16 .. +15 (contiguous)
    //   B: lane holds col (lane&15); K = hi16*16 .. +15 (contiguous)
    //   C: v8f element r -> row r + hi16*8, col lane&15
    const int nlo   = lane & 15;
    const int hi16  = lane >> 4;
    const int aklo  = hi16 * 8;
    const int crow0 = hi16 * 8;
    const int arowG = mg * 16 + nlo;          // A row within the 64-token tile

    const __bf16* xrow = xh + (size_t)(tok0 + arowG) * DDIM;

    v8f oacc[16];                             // persistent weighted output acc
    #pragma unroll
    for (int j = 0; j < 16; j++)
        #pragma unroll
        for (int q = 0; q < 8; q++) oacc[j][q] = 0.0f;

    for (int e = 0; e < NEXP; e++) {
        // Warm WGP$/L2 for next expert's weights while this one computes.
        if (e + 1 < NEXP) {
            __builtin_prefetch(W1t + (size_t)(e + 1) * HDIM * DDIM + (size_t)tid * 512, 0, 1);
            __builtin_prefetch(W2t + (size_t)(e + 1) * DDIM * HDIM + (size_t)tid * 512, 0, 1);
        }

        // ---- GEMM1: rows mg*16..+16, H-cols [ng*64, ng*64+64) ----------------
        v8f acc1[4];
        #pragma unroll
        for (int j = 0; j < 4; j++)
            #pragma unroll
            for (int q = 0; q < 8; q++) acc1[j][q] = 0.0f;

        const __bf16* W1e = W1t + (size_t)e * HDIM * DDIM;
        for (int kk = 0; kk < DDIM / 32; kk++) {
            const int k0 = kk * 32;
            AFrag a;
            a.h[0] = *(const v8bf*)&xrow[k0 + hi16 * 16];
            a.h[1] = *(const v8bf*)&xrow[k0 + hi16 * 16 + 8];
            #pragma unroll
            for (int j = 0; j < 4; j++) {
                const int n = ng * 64 + j * 16 + nlo;
                const v16bf bf = *(const v16bf*)&W1e[(size_t)n * DDIM + k0 + hi16 * 16];
                acc1[j] = __builtin_amdgcn_wmma_f32_16x16x32_bf16(
                    false, a.v, false, bf, (short)0, acc1[j], false, false);
            }
        }
        // bias + exact gelu, gate weight folded into h
        #pragma unroll
        for (int j = 0; j < 4; j++) {
            const int col = ng * 64 + j * 16 + nlo;
            const float b1v = b1[e * HDIM + col];
            #pragma unroll
            for (int r = 0; r < 8; r++) {
                const int row = mg * 16 + crow0 + r;
                float v = acc1[j][r] + b1v;
                float g = 0.5f * v * (1.0f + erff(v * 0.70710678118654752f));
                hs[row * HDIM + col] = (__bf16)(g * wls[row * NEXP + e]);
            }
        }
        __syncthreads();

        // ---- GEMM2: rows mg*16..+16, D-cols [ng*256, ng*256+256) -------------
        const __bf16* W2e = W2t + (size_t)e * DDIM * HDIM;
        const __bf16* hrow = &hs[arowG * HDIM];
        for (int kk = 0; kk < HDIM / 32; kk++) {
            const int k0 = kk * 32;
            AFrag a;
            a.h[0] = *(const v8bf*)&hrow[k0 + hi16 * 16];
            a.h[1] = *(const v8bf*)&hrow[k0 + hi16 * 16 + 8];
            #pragma unroll
            for (int j = 0; j < 16; j++) {
                const int dcol = ng * 256 + j * 16 + nlo;
                const v16bf bf = *(const v16bf*)&W2e[(size_t)dcol * HDIM + k0 + hi16 * 16];
                oacc[j] = __builtin_amdgcn_wmma_f32_16x16x32_bf16(
                    false, a.v, false, bf, (short)0, oacc[j], false, false);
            }
        }
        __syncthreads();   // hs is rewritten next expert iteration
    }

    // Epilogue: gated bias  sum_e w[row][e] * b2[e][dcol], then store fp32.
    #pragma unroll
    for (int j = 0; j < 16; j++) {
        const int dcol = ng * 256 + j * 16 + nlo;
        float b2v[NEXP];
        #pragma unroll
        for (int e = 0; e < NEXP; e++) b2v[e] = b2[e * DDIM + dcol];
        #pragma unroll
        for (int r = 0; r < 8; r++) {
            const int row = mg * 16 + crow0 + r;
            float bias = 0.0f;
            #pragma unroll
            for (int e = 0; e < NEXP; e++) bias += wls[row * NEXP + e] * b2v[e];
            out[(size_t)(tok0 + row) * DDIM + dcol] = oacc[j][r] + bias;
        }
    }
}

// ---------------------------------------------------------------------------
extern "C" void kernel_launch(void* const* d_in, const int* in_sizes, int n_in,
                              void* d_out, int out_size, void* d_ws, size_t ws_size,
                              hipStream_t stream) {
    (void)in_sizes; (void)n_in; (void)out_size; (void)ws_size;
    const float* x  = (const float*)d_in[0];
    const float* Wr = (const float*)d_in[1];
    const float* br = (const float*)d_in[2];
    const float* W1 = (const float*)d_in[3];
    const float* b1 = (const float*)d_in[4];
    const float* W2 = (const float*)d_in[5];
    const float* b2 = (const float*)d_in[6];
    float* out = (float*)d_out;

    char* ws = (char*)d_ws;
    float*  loadsum = (float*)(ws + 0);
    float*  wcomb   = (float*)(ws + 256);
    __bf16* W1t     = (__bf16*)(ws + 256 + (size_t)B_TOK * NEXP * sizeof(float));
    __bf16* W2t     = W1t + (size_t)NEXP * DDIM * HDIM;
    __bf16* xh      = W2t + (size_t)NEXP * DDIM * HDIM;

    init_kernel<<<1, 32, 0, stream>>>(loadsum);
    convert_x_kernel<<<(B_TOK * DDIM / 4 + 255) / 256, 256, 0, stream>>>(x, xh);
    convert_wt_kernel<<<(NEXP * DDIM * HDIM + 255) / 256, 256, 0, stream>>>(W1, W2, W1t, W2t);
    router_kernel<<<B_TOK / 8, 256, 0, stream>>>(x, Wr, br, wcomb, loadsum);
    aux_kernel<<<1, 32, 0, stream>>>(loadsum, out + (size_t)B_TOK * DDIM);
    moe_expert_kernel<<<B_TOK / 64, 512, 0, stream>>>(xh, W1t, b1, W2t, b2, wcomb, out);
}
